// SharedMatrix_38439957299746
// MI455X (gfx1250) — compile-verified, matrix operand order checked by reading, not run on hardware
//
#include <hip/hip_runtime.h>

typedef _Float16 v16h __attribute__((ext_vector_type(16)));
typedef _Float16 v8h  __attribute__((ext_vector_type(8)));
typedef float    v8f  __attribute__((ext_vector_type(8)));
typedef float    v4f  __attribute__((ext_vector_type(4)));

namespace {
constexpr int IN_F  = 4096;
constexpr int OUT_F = 4096;
constexpr int BATCH = 4096;
constexpr int BS    = 64;   // butterfly block size
constexpr int LRDIM = 64;   // low-rank dim
constexpr int A_ACT = 7;    // active blocks per input/output block (1 + log2(64))
constexpr int RS    = 72;   // padded LDS row stride in halfs (144B -> bank friendly)
constexpr int MT    = 256;  // batch rows per workgroup (fast path)

// ws layout (in halfs)
constexpr size_t WT16_OFF  = 0;                          // [64*7][64 n][64 k] f16
constexpr size_t L2T16_OFF = (size_t)64 * A_ACT * 4096;  // [64 p][64 n][64 k] f16
constexpr size_t TMP16_OFF = L2T16_OFF + (size_t)OUT_F / BS * 4096; // [4096][64] f16
constexpr size_t WS_HALFS  = TMP16_OFF + (size_t)BATCH * LRDIM;

// Async global->LDS 16-byte copy (ASYNCcnt tracked). IOFFSET applies to both
// the LDS destination and the global source address (ISA 10.x async pseudocode).
#define ASYNC_CP_B128(LDSOFF, VOFF, SBASE, IMM)                                \
  asm volatile("global_load_async_to_lds_b128 %0, %1, %2 offset:" IMM         \
               :: "v"(LDSOFF), "v"(VOFF), "s"(SBASE) : "memory")
#define WAIT_ASYNCCNT0() asm volatile("s_wait_asynccnt 0x0" ::: "memory")

__device__ __forceinline__ unsigned ldsAddr(const _Float16* p) {
  // Generic LDS pointer: high 32 bits = shared aperture, low 32 = LDS offset.
  return (unsigned)(unsigned long long)p;
}

// Load one 16x32 f16 WMMA fragment (A-rows or B-columns) from LDS.
__device__ __forceinline__ v16h ldsFrag(const _Float16* base, int row, int kb, int fh) {
  const _Float16* p = base + row * RS + kb + fh * 8;
  v8h lo = *(const v8h*)(p);
  v8h hi = *(const v8h*)(p + 16);
  v16h r;
#pragma unroll
  for (int i = 0; i < 8; ++i) { r[i] = lo[i]; r[i + 8] = hi[i]; }
  return r;
}

// Convert 16 contiguous f32 (global) -> 16 contiguous f16 (LDS or global).
__device__ __forceinline__ void cvtStore16(_Float16* d, const float* g) {
#pragma unroll
  for (int i = 0; i < 4; ++i) {
    v4f v = *(const v4f*)(g + i * 4);
    d[i * 4 + 0] = (_Float16)v[0];
    d[i * 4 + 1] = (_Float16)v[1];
    d[i * 4 + 2] = (_Float16)v[2];
    d[i * 4 + 3] = (_Float16)v[3];
  }
}
} // namespace

// ---- Preprocessing: straight f32 -> f16 convert (for lowrank_second) ----
__global__ __launch_bounds__(256)
void cvt16_kernel(const float* __restrict__ src, _Float16* __restrict__ dst) {
  const size_t i = ((size_t)blockIdx.x * 256 + threadIdx.x) * 16;
  cvtStore16(dst + i, src + i);
}

// ---- Preprocessing: transpose+convert one 64x64 weight block per WG ----
// wt16[(j*7+slot)*4096 + n*64 + k] = w[(j*64+k)*448 + slot*64 + n]
__global__ __launch_bounds__(256)
void cvtw_kernel(const float* __restrict__ w, _Float16* __restrict__ wt16) {
  __shared__ _Float16 T[64 * RS];  // T[k][n]
  const int j = blockIdx.x / A_ACT, slot = blockIdx.x % A_ACT;
  const int t = threadIdx.x;
  {
    const int k = t >> 2, q = t & 3;
    cvtStore16(T + k * RS + q * 16,
               w + (size_t)(j * BS + k) * (A_ACT * BS) + slot * BS + q * 16);
  }
  __syncthreads();
  {
    const int n = t >> 2, kq = t & 3;
    _Float16* d = wt16 + (size_t)(j * A_ACT + slot) * 4096 + n * BS + kq * 16;
#pragma unroll
    for (int e = 0; e < 16; ++e) d[e] = T[(kq * 16 + e) * RS + n];
  }
}

// ---- Kernel 1: tmp16[B,64] = x @ lowrank_first^T (f16 out) ----
__global__ __launch_bounds__(256)
void lr1_kernel(const float* __restrict__ x, const float* __restrict__ l1,
                _Float16* __restrict__ tmp) {
  __shared__ _Float16 As[64 * RS];
  __shared__ _Float16 Bs[64 * RS];
  const int t = threadIdx.x;
  const int lane = t & 31, wave = t >> 5;
  const int fr = lane & 15, fh = lane >> 4;
  const int row0 = blockIdx.x * 64;
  const int mt  = wave >> 1;
  const int nt0 = (wave & 1) * 2;
  const int sr = t >> 2, sq = t & 3;

  v8f acc0 = {}, acc1 = {};

  for (int kc = 0; kc < IN_F; kc += 64) {
    __syncthreads();
    cvtStore16(As + sr * RS + sq * 16, x  + (size_t)(row0 + sr) * IN_F + kc + sq * 16);
    cvtStore16(Bs + sr * RS + sq * 16, l1 + (size_t)sr * IN_F + kc + sq * 16);
    __syncthreads();
#pragma unroll
    for (int ks = 0; ks < 2; ++ks) {
      v16h a  = ldsFrag(As, mt * 16 + fr, ks * 32, fh);
      v16h b0 = ldsFrag(Bs, (nt0    ) * 16 + fr, ks * 32, fh);
      v16h b1 = ldsFrag(Bs, (nt0 + 1) * 16 + fr, ks * 32, fh);
      acc0 = __builtin_amdgcn_wmma_f32_16x16x32_f16(false, a, false, b0, (short)0, acc0, false, false);
      acc1 = __builtin_amdgcn_wmma_f32_16x16x32_f16(false, a, false, b1, (short)0, acc1, false, false);
    }
  }
#pragma unroll
  for (int v = 0; v < 8; ++v) {
    int r = row0 + mt * 16 + fh * 8 + v;
    tmp[(size_t)r * LRDIM + (nt0    ) * 16 + fr] = (_Float16)acc0[v];
    tmp[(size_t)r * LRDIM + (nt0 + 1) * 16 + fr] = (_Float16)acc1[v];
  }
}

// ---- Kernel 2 (fast path): 256x64 tile, 32x64 per wave, async + double buffer ----
__global__ __launch_bounds__(256)
void butterfly_kernel(const float* __restrict__ x, const _Float16* __restrict__ wt16,
                      const _Float16* __restrict__ l2t16, const _Float16* __restrict__ tmp,
                      const int* __restrict__ fidx, float* __restrict__ out) {
  __shared__ _Float16 As[2][MT * RS];  // 2 x 36 KB
  __shared__ _Float16 Bs[2][64 * RS];  // 2 x 9 KB
  const int t = threadIdx.x;
  const int lane = t & 31, wave = t >> 5;
  const int fr = lane & 15, fh = lane >> 4;
  const int row0 = blockIdx.x * MT;
  const int p = blockIdx.y;
  const int ar = t >> 1, ah = t & 1;  // A staging: row (pass adds 128), 32-half half
  const int bn = t >> 2, bq = t & 3;  // B staging: n row, 16-half quarter
  const int mrow = wave * 32;         // this wave's 32-row strip

  const unsigned voffB = (unsigned)((bn * BS + bq * 16) * 2);
  const unsigned voffA0 = (unsigned)(((row0 + ar      ) * LRDIM + ah * 32) * 2);
  const unsigned voffA1 = (unsigned)(((row0 + ar + 128) * LRDIM + ah * 32) * 2);
  const float* gx0 = x + (size_t)(row0 + ar      ) * IN_F + ah * 32;
  const float* gx1 = x + (size_t)(row0 + ar + 128) * IN_F + ah * 32;

  v8f acc[2][4] = {{{}, {}, {}, {}}, {{}, {}, {}, {}}};

  auto stage = [&](int s, int buf) {
    _Float16* Ad0 = &As[buf][(ar      ) * RS + ah * 32];
    _Float16* Ad1 = &As[buf][(ar + 128) * RS + ah * 32];
    const unsigned ldsB = ldsAddr(&Bs[buf][bn * RS + bq * 16]);
    if (s < A_ACT) {
      const int f = fidx[p * A_ACT + s];  // uniform
      const int j = f / A_ACT, slot = f % A_ACT;
      const _Float16* sB = wt16 + (size_t)(j * A_ACT + slot) * 4096;
      ASYNC_CP_B128(ldsB, voffB, sB, "0");
      ASYNC_CP_B128(ldsB, voffB, sB, "16");
      cvtStore16(Ad0,      gx0 + j * BS);
      cvtStore16(Ad0 + 16, gx0 + j * BS + 16);
      cvtStore16(Ad1,      gx1 + j * BS);
      cvtStore16(Ad1 + 16, gx1 + j * BS + 16);
    } else {
      const _Float16* sB = l2t16 + (size_t)p * 4096;
      ASYNC_CP_B128(ldsB, voffB, sB, "0");
      ASYNC_CP_B128(ldsB, voffB, sB, "16");
      const unsigned ldsA0 = ldsAddr(Ad0);
      ASYNC_CP_B128(ldsA0, voffA0, tmp, "0");
      ASYNC_CP_B128(ldsA0, voffA0, tmp, "16");
      ASYNC_CP_B128(ldsA0, voffA0, tmp, "32");
      ASYNC_CP_B128(ldsA0, voffA0, tmp, "48");
      const unsigned ldsA1 = ldsAddr(Ad1);
      ASYNC_CP_B128(ldsA1, voffA1, tmp, "0");
      ASYNC_CP_B128(ldsA1, voffA1, tmp, "16");
      ASYNC_CP_B128(ldsA1, voffA1, tmp, "32");
      ASYNC_CP_B128(ldsA1, voffA1, tmp, "48");
    }
  };

  stage(0, 0);
  WAIT_ASYNCCNT0();
  __syncthreads();

  for (int s = 0; s < 8; ++s) {
    const int buf = s & 1;
    if (s < 7) stage(s + 1, buf ^ 1);
    const _Float16* Ab = As[buf];
    const _Float16* Bb = Bs[buf];
#pragma unroll
    for (int ks = 0; ks < 2; ++ks) {
      v16h a0 = ldsFrag(Ab, mrow +      fr, ks * 32, fh);
      v16h a1 = ldsFrag(Ab, mrow + 16 + fr, ks * 32, fh);
#pragma unroll
      for (int nt = 0; nt < 4; ++nt) {
        v16h b = ldsFrag(Bb, nt * 16 + fr, ks * 32, fh);
        acc[0][nt] = __builtin_amdgcn_wmma_f32_16x16x32_f16(false, a0, false, b, (short)0, acc[0][nt], false, false);
        acc[1][nt] = __builtin_amdgcn_wmma_f32_16x16x32_f16(false, a1, false, b, (short)0, acc[1][nt], false, false);
      }
    }
    WAIT_ASYNCCNT0();
    __syncthreads();
  }

#pragma unroll
  for (int m = 0; m < 2; ++m) {
#pragma unroll
    for (int nt = 0; nt < 4; ++nt) {
#pragma unroll
      for (int v = 0; v < 8; ++v) {
        int r = row0 + mrow + m * 16 + fh * 8 + v;
        out[(size_t)r * OUT_F + p * BS + nt * 16 + fr] = acc[m][nt][v];
      }
    }
  }
}

// ---- Kernel 2 (fallback, small ws): fully inline conversion (round-1 path) ----
__global__ __launch_bounds__(256)
void butterfly_fb_kernel(const float* __restrict__ x, const float* __restrict__ w,
                         const float* __restrict__ l2, const _Float16* __restrict__ tmp,
                         const int* __restrict__ fidx, float* __restrict__ out) {
  __shared__ _Float16 As[128 * RS];
  __shared__ _Float16 Bs[64 * RS];
  const int t = threadIdx.x;
  const int lane = t & 31, wave = t >> 5;
  const int fr = lane & 15, fh = lane >> 4;
  const int row0 = blockIdx.x * 128;
  const int p = blockIdx.y;
  const int ar = t >> 1, ah = t & 1;
  const int br = t >> 2, bq = t & 3;

  v8f acc[4] = {{}, {}, {}, {}};

  for (int s = 0; s < 8; ++s) {
    __syncthreads();
    if (s < A_ACT) {
      const int f = fidx[p * A_ACT + s];
      const int j = f / A_ACT, slot = f % A_ACT;
      const float* gx = x + (size_t)(row0 + ar) * IN_F + j * BS + ah * 32;
      cvtStore16(As + ar * RS + ah * 32,      gx);
      cvtStore16(As + ar * RS + ah * 32 + 16, gx + 16);
      const float* gw = w + (size_t)(j * BS + br) * (A_ACT * BS) + slot * BS + bq * 16;
#pragma unroll
      for (int i = 0; i < 4; ++i) {
        v4f v = *(const v4f*)(gw + i * 4);
#pragma unroll
        for (int e = 0; e < 4; ++e)
          Bs[(bq * 16 + i * 4 + e) * RS + br] = (_Float16)v[e];
      }
    } else {
      const _Float16* gt = tmp + (size_t)(row0 + ar) * LRDIM + ah * 32;
      _Float16* d = As + ar * RS + ah * 32;
#pragma unroll
      for (int i = 0; i < 4; ++i) *(v8h*)(d + i * 8) = *(const v8h*)(gt + i * 8);
      cvtStore16(Bs + br * RS + bq * 16, l2 + (size_t)(p * BS + br) * LRDIM + bq * 16);
    }
    __syncthreads();
#pragma unroll
    for (int ks = 0; ks < 2; ++ks) {
      v16h a = ldsFrag(As, wave * 16 + fr, ks * 32, fh);
#pragma unroll
      for (int nt = 0; nt < 4; ++nt) {
        v16h b = ldsFrag(Bs, nt * 16 + fr, ks * 32, fh);
        acc[nt] = __builtin_amdgcn_wmma_f32_16x16x32_f16(false, a, false, b, (short)0, acc[nt], false, false);
      }
    }
  }
#pragma unroll
  for (int nt = 0; nt < 4; ++nt) {
#pragma unroll
    for (int v = 0; v < 8; ++v) {
      int r = row0 + wave * 16 + fh * 8 + v;
      out[(size_t)r * OUT_F + p * BS + nt * 16 + fr] = acc[nt][v];
    }
  }
}

extern "C" void kernel_launch(void* const* d_in, const int* in_sizes, int n_in,
                              void* d_out, int out_size, void* d_ws, size_t ws_size,
                              hipStream_t stream) {
  (void)in_sizes; (void)n_in; (void)out_size;
  const float* x    = (const float*)d_in[0];
  const float* w    = (const float*)d_in[1];
  const float* l1   = (const float*)d_in[2];
  const float* l2   = (const float*)d_in[3];
  const int*   fidx = (const int*)d_in[4];
  float* out = (float*)d_out;
  _Float16* ws = (_Float16*)d_ws;

  if (ws_size >= WS_HALFS * sizeof(_Float16)) {
    _Float16* wt16  = ws + WT16_OFF;
    _Float16* l2t16 = ws + L2T16_OFF;
    _Float16* tmp16 = ws + TMP16_OFF;
    cvtw_kernel<<<dim3(64 * A_ACT), 256, 0, stream>>>(w, wt16);
    cvt16_kernel<<<dim3((OUT_F * LRDIM) / (256 * 16)), 256, 0, stream>>>(l2, l2t16);
    lr1_kernel<<<dim3(BATCH / 64), 256, 0, stream>>>(x, l1, tmp16);
    butterfly_kernel<<<dim3(BATCH / MT, OUT_F / BS), 256, 0, stream>>>(
        x, wt16, l2t16, tmp16, fidx, out);
  } else {
    _Float16* tmp16 = ws;  // 512 KB
    lr1_kernel<<<dim3(BATCH / 64), 256, 0, stream>>>(x, l1, tmp16);
    butterfly_fb_kernel<<<dim3(BATCH / 128, OUT_F / BS), 256, 0, stream>>>(
        x, w, l2, tmp16, fidx, out);
  }
}
